// MixtureOfExperts_15607911153861
// MI455X (gfx1250) — matrix operand order repaired and checked by reading, not compiled, not run on hardware
//
#include <hip/hip_runtime.h>
#include <hip/hip_bf16.h>
#include <math.h>

typedef __bf16 bf16_t;
typedef __attribute__((ext_vector_type(16))) __bf16 v16bf;
typedef __attribute__((ext_vector_type(8)))  __bf16 v8bf;
typedef __attribute__((ext_vector_type(8)))  float  v8f;

typedef __attribute__((ext_vector_type(4))) unsigned int u32x4;
typedef __attribute__((ext_vector_type(8))) int          i32x8;
typedef __attribute__((ext_vector_type(4))) int          i32x4;

#define NTOK   20000
#define NEXP   8

// LDS layout (bytes), dynamic shared memory
#define A_TILE_BYTES   (64 * 40 * 2)            // 5120  : 64 rows x 32K, rows padded to 40
#define B_TILE_BYTES   (256 * 48 * 2)           // 24576 : 256 cols x 32K, rows padded to 48
#define A_TILE_ELEMS   (64 * 40)
#define B_TILE_ELEMS   (256 * 48)
#define H_BYTES        (64 * 264 * 2)           // 33792 : 64 x 256, rows padded to 264
#define SMEM_EXPERT    (2 * A_TILE_BYTES + 2 * B_TILE_BYTES + H_BYTES)   // 93184
#define SMEM_FINAL     (2 * A_TILE_BYTES + 2 * B_TILE_BYTES)            // 59392

#if defined(__AMDGCN__) && __has_builtin(__builtin_amdgcn_tensor_load_to_lds) && \
    __has_builtin(__builtin_amdgcn_s_wait_tensorcnt)
#define USE_TDM 1
#else
#define USE_TDM 0
#endif

// ---------------- small helpers ----------------
__device__ __forceinline__ v16bf mk16(v8bf lo, v8bf hi) {
  v16bf r;
#pragma unroll
  for (int i = 0; i < 8; ++i) { r[i] = lo[i]; r[i + 8] = hi[i]; }
  return r;
}
__device__ __forceinline__ v8f zero8f() {
  v8f r;
#pragma unroll
  for (int i = 0; i < 8; ++i) r[i] = 0.0f;
  return r;
}
__device__ __forceinline__ v8bf zero8bf() {
  v8bf r;
#pragma unroll
  for (int i = 0; i < 8; ++i) r[i] = (__bf16)0.0f;
  return r;
}
__device__ __forceinline__ float silu_f(float x) { return x / (1.0f + __expf(-x)); }

#if USE_TDM
// One TDM 2D tile load, Global -> LDS. Row-major bf16 tensor; global_addr is
// the TILE start. tensor_dim0 == tile_d0 (the K-window never exceeds the row,
// so no dim0-OOB is possible) -> group-1 descriptor is loop-invariant; only
// the global address changes per K-step. tensor_d1 = remaining rows (OOB rows
// are zero-filled, handling the M boundary). LDS gets (pad_amount_code+1)
// DWORDs of padding after every 16 DWORDs (= one 32-elem bf16 tile row).
__device__ __forceinline__ void tdm_load_2d(
    const void* gaddr, unsigned lds_addr,
    unsigned tensor_d1,
    unsigned tile_d0, unsigned tile_d1,
    unsigned stride_d0_elems,
    unsigned pad_interval_code, unsigned pad_amount_code) {
  unsigned long long ga = (unsigned long long)gaddr;
  u32x4 g0;
  g0[0] = 1u;                                          // count=1 (valid, user mode)
  g0[1] = lds_addr;                                    // LDS byte address
  g0[2] = (unsigned)(ga & 0xFFFFFFFFu);                // global_addr[31:0]
  g0[3] = (unsigned)((ga >> 32) & 0x01FFFFFFu)         // global_addr[56:32]
        | 0x80000000u;                                 // type=2 ("image")
  i32x8 g1;
  g1[0] = (int)((1u << 16)                             // data_size = 2 bytes
              | (1u << 20)                             // pad_enable
              | (pad_interval_code << 22)
              | (pad_amount_code << 25));
  g1[1] = (int)((tile_d0 & 0xFFFFu) << 16);            // tensor_dim0 = tile_d0
  g1[2] = (int)((tensor_d1 & 0xFFFFu) << 16);          // tensor_dim1[15:0]
  g1[3] = (int)(((tensor_d1 >> 16) & 0xFFFFu)          // tensor_dim1[31:16]
              | ((tile_d0 & 0xFFFFu) << 16));          // tile_dim0
  g1[4] = (int)(tile_d1 & 0xFFFFu);                    // tile_dim1 (tile_dim2=0)
  g1[5] = (int)stride_d0_elems;                        // tensor_dim0_stride[31:0]
  g1[6] = 0;
  g1[7] = 0;
  i32x4 z4 = {0, 0, 0, 0};
  i32x8 z8 = {0, 0, 0, 0, 0, 0, 0, 0};
  __builtin_amdgcn_tensor_load_to_lds(g0, g1, z4, z4, z8, 0);
}
__device__ __forceinline__ unsigned lds_addr_of(const void* p) {
  // addr[31:0] of a generic LDS pointer is the LDS byte offset; pin to SGPR.
  return (unsigned)__builtin_amdgcn_readfirstlane((int)(unsigned long long)p);
}
#endif

// ---------------- prep kernels ----------------
__global__ __launch_bounds__(256) void prep_scalar_kernel(
    const float* __restrict__ ss, const float* __restrict__ sl,
    const float* __restrict__ vs, const float* __restrict__ vl,
    bf16_t* __restrict__ sc, bf16_t* __restrict__ vp, int N) {
  int i = blockIdx.x * 256 + threadIdx.x;
  if (i >= N * 512) return;
  int n = i >> 9, d = i & 511;
  float s = (d < 256) ? ss[n * 256 + d] : sl[n * 256 + (d - 256)];
  sc[i] = (bf16_t)s;
  float a, b, c;
  if (d < 256) {
    a = vs[(n * 3 + 0) * 256 + d];
    b = vs[(n * 3 + 1) * 256 + d];
    c = vs[(n * 3 + 2) * 256 + d];
  } else {
    int dd = d - 256;
    a = vl[(n * 3 + 0) * 256 + dd];
    b = vl[(n * 3 + 1) * 256 + dd];
    c = vl[(n * 3 + 2) * 256 + dd];
  }
  vp[i] = (bf16_t)((a + b + c) * (1.0f / 3.0f));
}

__global__ __launch_bounds__(256) void prep_vector_kernel(
    const float* __restrict__ vs, const float* __restrict__ vl,
    bf16_t* __restrict__ vc, int M3) {
  int i = blockIdx.x * 256 + threadIdx.x;
  if (i >= M3 * 512) return;
  int r = i >> 9, d = i & 511;
  float v = (d < 256) ? vs[r * 256 + d] : vl[r * 256 + (d - 256)];
  vc[i] = (bf16_t)v;
}

__global__ __launch_bounds__(256) void transpose_to_bf16_kernel(
    const float* __restrict__ src, bf16_t* __restrict__ dst,
    int R, int C, int total) {
  int i = blockIdx.x * 256 + threadIdx.x;
  if (i >= total) return;
  int rc = R * C;
  int b = i / rc, rem = i - b * rc;
  int r = rem / C, c = rem - r * C;
  dst[b * rc + c * R + r] = (bf16_t)src[i];
}

// ---------------- gate kernel (VALU, <2% of FLOPs) ----------------
__global__ __launch_bounds__(256) void gate_kernel(
    const float* __restrict__ ss, const float* __restrict__ sl,
    const float* __restrict__ vs, const float* __restrict__ vl,
    const float* __restrict__ Wg1, const float* __restrict__ bg1,
    const float* __restrict__ Wg2, const float* __restrict__ bg2,
    float* __restrict__ w_s, float* __restrict__ w_v, int N) {
  __shared__ float Xs[8][512];
  __shared__ float Hsh[8][128];
  const int wave = threadIdx.x >> 5, lane = threadIdx.x & 31;
  const int T = blockIdx.x * 8 + wave;
  const bool isV = (T >= N);
  const int n = isV ? (T - N) : T;

#pragma unroll
  for (int c = 0; c < 16; ++c) {
    int d = c * 32 + lane;
    float x;
    if (!isV) {
      x = (d < 256) ? ss[n * 256 + d] : sl[n * 256 + (d - 256)];
    } else if (d < 256) {
      x = (vs[(n * 3 + 0) * 256 + d] + vs[(n * 3 + 1) * 256 + d] +
           vs[(n * 3 + 2) * 256 + d]) * (1.0f / 3.0f);
    } else {
      int dd = d - 256;
      x = (vl[(n * 3 + 0) * 256 + dd] + vl[(n * 3 + 1) * 256 + dd] +
           vl[(n * 3 + 2) * 256 + dd]) * (1.0f / 3.0f);
    }
    Xs[wave][d] = x;
  }
  __syncthreads();

  float acc[4];
#pragma unroll
  for (int c = 0; c < 4; ++c) acc[c] = bg1[lane + 32 * c];
  for (int d = 0; d < 512; ++d) {
    float xv = Xs[wave][d];
    const float* wr = Wg1 + d * 128;
#pragma unroll
    for (int c = 0; c < 4; ++c) acc[c] += xv * wr[lane + 32 * c];
  }
#pragma unroll
  for (int c = 0; c < 4; ++c) Hsh[wave][lane + 32 * c] = silu_f(acc[c]);
  __syncthreads();

  if (lane < 8) {
    float a = bg2[lane];
#pragma unroll 4
    for (int j = 0; j < 128; ++j) a += Hsh[wave][j] * Wg2[j * 8 + lane];
    float m = a;
    for (int o = 1; o < 8; o <<= 1) m = fmaxf(m, __shfl_xor(m, o, 8));
    float ex = __expf(a - m);
    float s = ex;
    for (int o = 1; o < 8; o <<= 1) s += __shfl_xor(s, o, 8);
    float* dst = isV ? w_v : w_s;
    dst[n * 8 + lane] = ex / s;
  }
}

// ---------------- fused expert kernel (TDM double-buffered) ----------------
template <bool HAS_BIAS, bool VEC_GATE>
__global__ __launch_bounds__(256) void expert_fused_kernel(
    const bf16_t* __restrict__ X,    // [M][512]
    const bf16_t* __restrict__ W1t,  // [E][256][512]  (out-major, K contiguous)
    const float*  __restrict__ b1,   // [E][256] or null
    const bf16_t* __restrict__ W2t,  // [E][256][256]
    const float*  __restrict__ b2,   // [E][256] or null
    const float*  __restrict__ gate, // [Ng][8]
    bf16_t* __restrict__ mid,        // [M][256]
    int M) {
  extern __shared__ char smem[];
  bf16_t* As = (bf16_t*)smem;                                        // [2][64*40]
  bf16_t* Bs = (bf16_t*)(smem + 2 * A_TILE_BYTES);                   // [2][256*48]
  bf16_t* Hs = (bf16_t*)(smem + 2 * A_TILE_BYTES + 2 * B_TILE_BYTES);// [64*264]

  const int tid  = threadIdx.x;
  const int lane = tid & 31;
  const int wave = tid >> 5;
  const int wm   = wave >> 1;  // 0..3 : 16-row slab
  const int wn   = wave & 1;   // 0..1 : 128-col half
  const int ml   = lane & 15;
  const int hi   = lane >> 4;
  const int row0 = blockIdx.x * 64;
  // scalar (SGPR) leader predicate -> s_cbranch, no EXEC save/restore
  const bool leader = (__builtin_amdgcn_readfirstlane((int)threadIdx.x) >> 5) == 0;

#if USE_TDM
  const unsigned ldsA0 = lds_addr_of(As);
  const unsigned ldsB0 = lds_addr_of(Bs);
#endif

  v8f midacc[8];
#pragma unroll
  for (int t = 0; t < 8; ++t) midacc[t] = zero8f();

  for (int e = 0; e < NEXP; ++e) {
    v8f acc[8];
#pragma unroll
    for (int t = 0; t < 8; ++t) acc[t] = zero8f();

    // stage tile kk of layer 1 (A: X rows, B: W1t[e] cols) into buffer `buf`
    auto stage1 = [&](int kk, int buf) {
#if USE_TDM
      if (leader) {
        tdm_load_2d(X + (size_t)row0 * 512 + kk * 32,
                    ldsA0 + (unsigned)buf * A_TILE_BYTES,
                    (unsigned)(M - row0), 32, 64, 512,
                    /*interval 16dw*/ 3, /*pad 4dw*/ 3);
        tdm_load_2d(W1t + (size_t)e * 256 * 512 + kk * 32,
                    ldsB0 + (unsigned)buf * B_TILE_BYTES,
                    256, 32, 256, 512,
                    /*interval 16dw*/ 3, /*pad 8dw*/ 7);
      }
#else
      {
        int r = tid >> 2, c = (tid & 3) * 8;
        int grow = row0 + r;
        v8bf v = zero8bf();
        if (grow < M) v = *(const v8bf*)(X + (size_t)grow * 512 + kk * 32 + c);
        *(v8bf*)(As + buf * A_TILE_ELEMS + r * 40 + c) = v;
      }
      {
        const bf16_t* src = W1t + (size_t)(e * 256 + tid) * 512 + kk * 32;
#pragma unroll
        for (int c = 0; c < 4; ++c)
          *(v8bf*)(Bs + buf * B_TILE_ELEMS + tid * 48 + c * 8) =
              *(const v8bf*)(src + c * 8);
      }
#endif
    };

    // -------- layer 1 : K = 512 (16 steps of 32), ping-pong --------
    stage1(0, 0);
    for (int kk = 0; kk < 16; ++kk) {
      const int cur = kk & 1;
      if (kk + 1 < 16) stage1(kk + 1, cur ^ 1);
#if USE_TDM
      if (kk + 1 < 16) __builtin_amdgcn_s_wait_tensorcnt(2);  // next stage in flight
      else             __builtin_amdgcn_s_wait_tensorcnt(0);
#endif
      __syncthreads();
      const bf16_t* Ab = As + cur * A_TILE_ELEMS;
      const bf16_t* Bb = Bs + cur * B_TILE_ELEMS;
      const bf16_t* arow = Ab + (wm * 16 + ml) * 40;
      v16bf afrag = mk16(*(const v8bf*)(arow + 16 * hi),
                         *(const v8bf*)(arow + 16 * hi + 8));
#pragma unroll
      for (int t = 0; t < 8; ++t) {
        const bf16_t* brow = Bb + (wn * 128 + t * 16 + ml) * 48 + 16 * hi;
        v16bf bfrag = mk16(*(const v8bf*)brow, *(const v8bf*)(brow + 8));
        acc[t] = __builtin_amdgcn_wmma_f32_16x16x32_bf16(
            false, afrag, false, bfrag, (short)0, acc[t], false, false);
      }
      __syncthreads();
    }

    // epilogue 1: bias + SiLU -> Hs (bf16)
#pragma unroll
    for (int t = 0; t < 8; ++t) {
      const int col = wn * 128 + t * 16 + ml;
      const float bv = HAS_BIAS ? b1[e * 256 + col] : 0.0f;
#pragma unroll
      for (int v = 0; v < 8; ++v) {
        float x = acc[t][v] + bv;
        Hs[(wm * 16 + v + 8 * hi) * 264 + col] = (bf16_t)silu_f(x);
      }
      acc[t] = zero8f();
    }

    // stage tile kk of layer 2 (B only: W2t[e]) into buffer `buf`
    auto stage2 = [&](int kk, int buf) {
#if USE_TDM
      if (leader) {
        tdm_load_2d(W2t + (size_t)e * 256 * 256 + kk * 32,
                    ldsB0 + (unsigned)buf * B_TILE_BYTES,
                    256, 32, 256, 256, 3, 7);
      }
#else
      const bf16_t* src = W2t + (size_t)(e * 256 + tid) * 256 + kk * 32;
#pragma unroll
      for (int c = 0; c < 4; ++c)
        *(v8bf*)(Bs + buf * B_TILE_ELEMS + tid * 48 + c * 8) =
            *(const v8bf*)(src + c * 8);
#endif
    };

    // -------- layer 2 : K = 256 (8 steps of 32), ping-pong --------
    stage2(0, 0);
    for (int kk = 0; kk < 8; ++kk) {
      const int cur = kk & 1;
      if (kk + 1 < 8) stage2(kk + 1, cur ^ 1);
#if USE_TDM
      if (kk + 1 < 8) __builtin_amdgcn_s_wait_tensorcnt(1);
      else            __builtin_amdgcn_s_wait_tensorcnt(0);
#endif
      __syncthreads();
      const bf16_t* Bb = Bs + cur * B_TILE_ELEMS;
      const bf16_t* arow = Hs + (wm * 16 + ml) * 264 + kk * 32;
      v16bf afrag = mk16(*(const v8bf*)(arow + 16 * hi),
                         *(const v8bf*)(arow + 16 * hi + 8));
#pragma unroll
      for (int t = 0; t < 8; ++t) {
        const bf16_t* brow = Bb + (wn * 128 + t * 16 + ml) * 48 + 16 * hi;
        v16bf bfrag = mk16(*(const v8bf*)brow, *(const v8bf*)(brow + 8));
        acc[t] = __builtin_amdgcn_wmma_f32_16x16x32_bf16(
            false, afrag, false, bfrag, (short)0, acc[t], false, false);
      }
      __syncthreads();
    }

    // epilogue 2: bias + gate-weighted accumulate
    float gw[8];
#pragma unroll
    for (int v = 0; v < 8; ++v) {
      int row = row0 + wm * 16 + v + 8 * hi;
      float g = 0.0f;
      if (row < M) {
        int gr = VEC_GATE ? (row / 3) : row;
        g = gate[gr * 8 + e];
      }
      gw[v] = g;
    }
#pragma unroll
    for (int t = 0; t < 8; ++t) {
      const int col = wn * 128 + t * 16 + ml;
      const float bv = HAS_BIAS ? b2[e * 256 + col] : 0.0f;
#pragma unroll
      for (int v = 0; v < 8; ++v)
        midacc[t][v] += gw[v] * (acc[t][v] + bv);
    }
  }

  // store mixed result (bf16)
#pragma unroll
  for (int t = 0; t < 8; ++t) {
    const int col = wn * 128 + t * 16 + ml;
#pragma unroll
    for (int v = 0; v < 8; ++v) {
      int row = row0 + wm * 16 + v + 8 * hi;
      if (row < M) mid[(size_t)row * 256 + col] = (bf16_t)midacc[t][v];
    }
  }
}

// ---------------- final projection : out = mid @ Wf (+ bias), fp32 out ----------------
template <bool HAS_BIAS>
__global__ __launch_bounds__(256) void final_proj_kernel(
    const bf16_t* __restrict__ Xm,   // [M][256]
    const bf16_t* __restrict__ Wt,   // [256][256] (out-major)
    const float*  __restrict__ bias, // [256] or null
    float* __restrict__ out,         // [M][256]
    int M) {
  extern __shared__ char smem[];
  bf16_t* As = (bf16_t*)smem;                       // [2][64*40]
  bf16_t* Bs = (bf16_t*)(smem + 2 * A_TILE_BYTES);  // [2][256*48]

  const int tid  = threadIdx.x;
  const int lane = tid & 31;
  const int wave = tid >> 5;
  const int wm   = wave >> 1;
  const int wn   = wave & 1;
  const int ml   = lane & 15;
  const int hi   = lane >> 4;
  const int row0 = blockIdx.x * 64;
  const bool leader = (__builtin_amdgcn_readfirstlane((int)threadIdx.x) >> 5) == 0;

#if USE_TDM
  const unsigned ldsA0 = lds_addr_of(As);
  const unsigned ldsB0 = lds_addr_of(Bs);
#endif

  v8f acc[8];
#pragma unroll
  for (int t = 0; t < 8; ++t) acc[t] = zero8f();

  auto stage = [&](int kk, int buf) {
#if USE_TDM
    if (leader) {
      tdm_load_2d(Xm + (size_t)row0 * 256 + kk * 32,
                  ldsA0 + (unsigned)buf * A_TILE_BYTES,
                  (unsigned)(M - row0), 32, 64, 256, 3, 3);
      tdm_load_2d(Wt + kk * 32,
                  ldsB0 + (unsigned)buf * B_TILE_BYTES,
                  256, 32, 256, 256, 3, 7);
    }
#else
    {
      int r = tid >> 2, c = (tid & 3) * 8;
      int grow = row0 + r;
      v8bf v = zero8bf();
      if (grow < M) v = *(const v8bf*)(Xm + (size_t)grow * 256 + kk * 32 + c);
      *(v8bf*)(As + buf * A_TILE_ELEMS + r * 40 + c) = v;
    }
    {
      const bf16_t* src = Wt + (size_t)tid * 256 + kk * 32;
#pragma unroll
      for (int c = 0; c < 4; ++c)
        *(v8bf*)(Bs + buf * B_TILE_ELEMS + tid * 48 + c * 8) =
            *(const v8bf*)(src + c * 8);
    }
#endif
  };

  stage(0, 0);
  for (int kk = 0; kk < 8; ++kk) {
    const int cur = kk & 1;
    if (kk + 1 < 8) stage(kk + 1, cur ^ 1);
#if USE_TDM
    if (kk + 1 < 8) __builtin_amdgcn_s_wait_tensorcnt(2);
    else            __builtin_amdgcn_s_wait_tensorcnt(0);
#endif
    __syncthreads();
    const bf16_t* Ab = As + cur * A_TILE_ELEMS;
    const bf16_t* Bb = Bs + cur * B_TILE_ELEMS;
    const bf16_t* arow = Ab + (wm * 16 + ml) * 40;
    v16bf afrag = mk16(*(const v8bf*)(arow + 16 * hi),
                       *(const v8bf*)(arow + 16 * hi + 8));
#pragma unroll
    for (int t = 0; t < 8; ++t) {
      const bf16_t* brow = Bb + (wn * 128 + t * 16 + ml) * 48 + 16 * hi;
      v16bf bfrag = mk16(*(const v8bf*)brow, *(const v8bf*)(brow + 8));
      acc[t] = __builtin_amdgcn_wmma_f32_16x16x32_bf16(
          false, afrag, false, bfrag, (short)0, acc[t], false, false);
    }
    __syncthreads();
  }

#pragma unroll
  for (int t = 0; t < 8; ++t) {
    const int col = wn * 128 + t * 16 + ml;
    const float bv = HAS_BIAS ? bias[col] : 0.0f;
#pragma unroll
    for (int v = 0; v < 8; ++v) {
      int row = row0 + wm * 16 + v + 8 * hi;
      if (row < M) out[(size_t)row * 256 + col] = acc[t][v] + bv;
    }
  }
}

// ---------------- launcher ----------------
static inline void* bump(char*& p, size_t bytes) {
  void* r = (void*)p;
  p += (bytes + 255) & ~(size_t)255;
  return r;
}

extern "C" void kernel_launch(void* const* d_in, const int* in_sizes, int n_in,
                              void* d_out, int out_size, void* d_ws, size_t ws_size,
                              hipStream_t stream) {
  const float* ss  = (const float*)d_in[0];
  const float* sl  = (const float*)d_in[1];
  const float* vs  = (const float*)d_in[2];
  const float* vl  = (const float*)d_in[3];
  const float* Wg1 = (const float*)d_in[4];
  const float* bg1 = (const float*)d_in[5];
  const float* Wg2 = (const float*)d_in[6];
  const float* bg2 = (const float*)d_in[7];
  const float* Ws1 = (const float*)d_in[8];
  const float* bs1 = (const float*)d_in[9];
  const float* Ws2 = (const float*)d_in[10];
  const float* bs2 = (const float*)d_in[11];
  const float* Wv1 = (const float*)d_in[12];
  const float* Wv2 = (const float*)d_in[13];
  const float* Wfs = (const float*)d_in[14];
  const float* bfs = (const float*)d_in[15];
  const float* Wfv = (const float*)d_in[16];

  const int N = NTOK, M3 = 3 * NTOK;
  float* out_s = (float*)d_out;
  float* out_v = out_s + (size_t)N * 256;

  char* p = (char*)d_ws;
  bf16_t* sc   = (bf16_t*)bump(p, (size_t)N  * 512 * sizeof(bf16_t));
  bf16_t* vc   = (bf16_t*)bump(p, (size_t)M3 * 512 * sizeof(bf16_t));
  bf16_t* vp   = (bf16_t*)bump(p, (size_t)N  * 512 * sizeof(bf16_t));
  bf16_t* Ws1t = (bf16_t*)bump(p, (size_t)8 * 512 * 256 * sizeof(bf16_t));
  bf16_t* Ws2t = (bf16_t*)bump(p, (size_t)8 * 256 * 256 * sizeof(bf16_t));
  bf16_t* Wv1t = (bf16_t*)bump(p, (size_t)8 * 512 * 256 * sizeof(bf16_t));
  bf16_t* Wv2t = (bf16_t*)bump(p, (size_t)8 * 256 * 256 * sizeof(bf16_t));
  bf16_t* Wfst = (bf16_t*)bump(p, (size_t)256 * 256 * sizeof(bf16_t));
  bf16_t* Wfvt = (bf16_t*)bump(p, (size_t)256 * 256 * sizeof(bf16_t));
  float*  w_s  = (float*)bump(p, (size_t)N * 8 * sizeof(float));
  float*  w_v  = (float*)bump(p, (size_t)N * 8 * sizeof(float));
  bf16_t* mids = (bf16_t*)bump(p, (size_t)N  * 256 * sizeof(bf16_t));
  bf16_t* midv = (bf16_t*)bump(p, (size_t)M3 * 256 * sizeof(bf16_t));

  // Allow >64KB dynamic LDS (WGP has 320KB). Deterministic, not a stream op.
  (void)hipFuncSetAttribute(reinterpret_cast<const void*>(&expert_fused_kernel<true, false>),
                            hipFuncAttributeMaxDynamicSharedMemorySize, SMEM_EXPERT);
  (void)hipFuncSetAttribute(reinterpret_cast<const void*>(&expert_fused_kernel<false, true>),
                            hipFuncAttributeMaxDynamicSharedMemorySize, SMEM_EXPERT);
  (void)hipFuncSetAttribute(reinterpret_cast<const void*>(&final_proj_kernel<true>),
                            hipFuncAttributeMaxDynamicSharedMemorySize, SMEM_FINAL);
  (void)hipFuncSetAttribute(reinterpret_cast<const void*>(&final_proj_kernel<false>),
                            hipFuncAttributeMaxDynamicSharedMemorySize, SMEM_FINAL);

  prep_scalar_kernel<<<(N * 512 + 255) / 256, 256, 0, stream>>>(ss, sl, vs, vl, sc, vp, N);
  prep_vector_kernel<<<(M3 * 512 + 255) / 256, 256, 0, stream>>>(vs, vl, vc, M3);

  transpose_to_bf16_kernel<<<(8 * 512 * 256 + 255) / 256, 256, 0, stream>>>(Ws1, Ws1t, 512, 256, 8 * 512 * 256);
  transpose_to_bf16_kernel<<<(8 * 256 * 256 + 255) / 256, 256, 0, stream>>>(Ws2, Ws2t, 256, 256, 8 * 256 * 256);
  transpose_to_bf16_kernel<<<(8 * 512 * 256 + 255) / 256, 256, 0, stream>>>(Wv1, Wv1t, 512, 256, 8 * 512 * 256);
  transpose_to_bf16_kernel<<<(8 * 256 * 256 + 255) / 256, 256, 0, stream>>>(Wv2, Wv2t, 256, 256, 8 * 256 * 256);
  transpose_to_bf16_kernel<<<(256 * 256 + 255) / 256, 256, 0, stream>>>(Wfs, Wfst, 256, 256, 256 * 256);
  transpose_to_bf16_kernel<<<(256 * 256 + 255) / 256, 256, 0, stream>>>(Wfv, Wfvt, 256, 256, 256 * 256);

  gate_kernel<<<(2 * N) / 8, 256, 0, stream>>>(ss, sl, vs, vl, Wg1, bg1, Wg2, bg2, w_s, w_v, N);

  expert_fused_kernel<true, false><<<(N + 63) / 64, 256, SMEM_EXPERT, stream>>>(
      sc, Ws1t, bs1, Ws2t, bs2, w_s, mids, N);
  expert_fused_kernel<false, true><<<(M3 + 63) / 64, 256, SMEM_EXPERT, stream>>>(
      vc, Wv1t, nullptr, Wv2t, nullptr, w_v, midv, M3);

  final_proj_kernel<true><<<(N + 63) / 64, 256, SMEM_FINAL, stream>>>(mids, Wfst, bfs, out_s, N);
  final_proj_kernel<false><<<(M3 + 63) / 64, 256, SMEM_FINAL, stream>>>(midv, Wfvt, nullptr, out_v, M3);
}
